// MemoryBankV2_13056700580081
// MI455X (gfx1250) — matrix operand order repaired and checked by reading, not compiled, not run on hardware
//
#include <hip/hip_runtime.h>
#include <hip/hip_bf16.h>

typedef _Float16     v16h  __attribute__((ext_vector_type(16)));
typedef float        v8f   __attribute__((ext_vector_type(8)));
typedef unsigned int u32x4 __attribute__((ext_vector_type(4)));
typedef int          i32x4 __attribute__((ext_vector_type(4)));
typedef int          i32x8 __attribute__((ext_vector_type(8)));

#define Bb 128
#define Tt 32
#define Dd 512
#define Ll 2
#define Mrows (Bb*Tt)   // 4096

// ---------------------------------------------------------------------------
// f32 -> f16 convert
// ---------------------------------------------------------------------------
__global__ __launch_bounds__(256) void cvt_f16_kernel(const float* __restrict__ src,
                                                      _Float16* __restrict__ dst) {
    int i = blockIdx.x * 256 + threadIdx.x;
    dst[i] = (_Float16)src[i];
}

// ---------------------------------------------------------------------------
// TDM: async 2D f16 tile load (rows x 32 halves) global -> LDS.
// LDS row stride is padded to 40 halves by the TDM pad feature:
//   row = 32 halves = 16 dwords -> pad_interval=3 (16 dw), pad_amount=3 (+4 dw)
// Descriptor per CDNA5 ISA ch.8 (D# group0/group1). Wave-level op, EXEC ignored.
// ---------------------------------------------------------------------------
__device__ __forceinline__ void tdm_load_tile_f16(unsigned ldsOff, const _Float16* gsrc,
                                                  int rowStride, int totRows,
                                                  int tileRows) {
    const unsigned long long ga = (unsigned long long)(uintptr_t)gsrc;
    u32x4 g0;
    g0[0] = 1u;                                   // count=1, user descriptor
    g0[1] = ldsOff;                               // lds_addr (bytes)
    g0[2] = (unsigned)ga;                         // global_addr[31:0]
    g0[3] = (unsigned)(ga >> 32) | 0x80000000u;   // global_addr[56:32] | type=2
    i32x8 g1;
    g1[0] = (int)((1u << 16)      // data_size = 2 bytes (f16)
                | (1u << 20)      // pad_enable
                | (3u << 22)      // pad_interval: 16 dwords
                | (3u << 25));    // pad_amount : 4 dwords (+16B -> 40-half stride)
    g1[1] = (int)(((unsigned)rowStride & 0xFFFFu) << 16);            // tensor_dim0 lo
    g1[2] = (int)(((unsigned)rowStride >> 16)
                | (((unsigned)totRows & 0xFFFFu) << 16));            // dim0 hi | dim1 lo
    g1[3] = (int)(((unsigned)totRows >> 16) | (32u << 16));          // dim1 hi | tile_dim0=32
    g1[4] = (int)(unsigned)tileRows;                                 // tile_dim1 (tile_dim2=0)
    g1[5] = rowStride;                                               // tensor_dim0_stride
    g1[6] = 0;
    g1[7] = 0;
    i32x4 z4 = {0, 0, 0, 0};
#if __clang_major__ >= 23
    i32x8 z8 = {0, 0, 0, 0, 0, 0, 0, 0};
    __builtin_amdgcn_tensor_load_to_lds(g0, g1, z4, z4, z8, 0);
#else
    __builtin_amdgcn_tensor_load_to_lds(g0, g1, z4, z4, 0);
#endif
}

// ---------------------------------------------------------------------------
// Fused epilogue store for one 16x16 f32 accumulator tile.
// C layout: n = lane&15, VGPR j -> m = j + 8*(lane>>4).
// ---------------------------------------------------------------------------
__device__ __forceinline__ void store_tile(v8f acc, int m0, int n0, int lane,
                                           const float* __restrict__ bias, float alpha,
                                           int applyMask, int actGelu, int accum,
                                           float* __restrict__ outF32,
                                           _Float16* __restrict__ outF16, int ldc) {
    const int h   = lane >> 4;
    const int n_g = n0 + (lane & 15);
#pragma unroll
    for (int j = 0; j < 8; ++j) {
        const int m_g = m0 + h * 8 + j;
        float v = acc[j] * alpha;
        if (bias) v += bias[n_g];
        if (actGelu) v = 0.5f * v * (1.0f + erff(v * 0.70710678f));
        if (applyMask && ((n_g >> 5) >= (m_g >> 5))) v = -60000.0f;   // item visibility
        const size_t idx = (size_t)m_g * ldc + n_g;
        if (outF32) { float r = v; if (accum) r += outF32[idx]; outF32[idx] = r; }
        if (outF16) outF16[idx] = (_Float16)v;
    }
}

// ---------------------------------------------------------------------------
// Generic tiled WMMA GEMM:
//   C[M,N] = act( alpha * A[M,K] @ op(B) + bias ), optional mask / accumulate
//   BMODE 0: B stored [N,K] row-major (multiply by B^T)  -- weights, Q@K^T
//   BMODE 1: B stored [K,N] row-major                    -- probs @ V
// Block: 256 thr (8 wave32). Block tile 128x64, wave tile 32x32, K step 32.
// Double-buffered LDS; tiles streamed by the Tensor Data Mover one step ahead
// (s_wait_tensorcnt <= ops-in-flight) so DMA overlaps the WMMA pipeline.
// ---------------------------------------------------------------------------
template<int BMODE>
__global__ __launch_bounds__(256)
void gemm_wmma_kernel(const _Float16* __restrict__ A, int lda,
                      const _Float16* __restrict__ Bm, int ldb,
                      const float* __restrict__ bias,
                      float alpha, int K,
                      int applyMask, int actGelu, int accum,
                      float* __restrict__ outF32, _Float16* __restrict__ outF16,
                      int ldc) {
    __shared__ _Float16 As[2][128 * 40];   // 32 K + 8 pad per row
    __shared__ _Float16 Bs[2][64 * 40];    // stored [n][k]

    const int tid  = threadIdx.x;
    const int lane = tid & 31;
    const int wave = tid >> 5;
    const int wm   = (wave >> 1) * 32;   // wave row offset in block tile (0..96)
    const int wn   = (wave & 1) * 32;    // wave col offset in block tile (0 or 32)
    const int mBlock = blockIdx.y * 128;
    const int nBlock = blockIdx.x * 64;
    const int Mtot = gridDim.y * 128;
    const int Ntot = gridDim.x * 64;

    // staging indices for BMODE==1 B transpose: 32 k x 64 n, 8 n/thread
    const int bk = tid >> 3;
    const int bn = (tid & 7) * 8;

    const unsigned asOff0 = (unsigned)(uintptr_t)&As[0][0];
    const unsigned bsOff0 = (unsigned)(uintptr_t)&Bs[0][0];
    const unsigned asStep = 128 * 40 * 2;   // bytes per A buffer
    const unsigned bsStep = 64 * 40 * 2;    // bytes per B buffer

    v8f acc00 = {}, acc01 = {}, acc10 = {}, acc11 = {};

    const int nk = K >> 5;

    // ---- prologue: stage step 0 into buffer 0 ----
    if (BMODE == 1) {
        const _Float16* src = Bm + (size_t)bk * ldb + (nBlock + bn);
        float4 t = *(const float4*)src;
        const _Float16* tp = (const _Float16*)&t;
#pragma unroll
        for (int j = 0; j < 8; ++j) Bs[0][(bn + j) * 40 + bk] = tp[j];
    }
    if (wave == 0) {
        tdm_load_tile_f16(asOff0, A + (size_t)mBlock * lda, lda, Mtot, 128);
        if (BMODE == 0)
            tdm_load_tile_f16(bsOff0, Bm + (size_t)nBlock * ldb, ldb, Ntot, 64);
    }

    for (int ks = 0; ks < nk; ++ks) {
        const int cur = ks & 1;
        const int nxt = cur ^ 1;
        // ---- stage step ks+1 into the alternate buffer (async) ----
        if (ks + 1 < nk) {
            const int k1 = (ks + 1) << 5;
            if (BMODE == 1) {
                const _Float16* src = Bm + (size_t)(k1 + bk) * ldb + (nBlock + bn);
                float4 t = *(const float4*)src;
                const _Float16* tp = (const _Float16*)&t;
#pragma unroll
                for (int j = 0; j < 8; ++j) Bs[nxt][(bn + j) * 40 + bk] = tp[j];
                if (k1 + 32 < K)
                    __builtin_prefetch(src + 32 * ldb, 0, 1);
            }
            if (wave == 0) {
                tdm_load_tile_f16(asOff0 + (unsigned)nxt * asStep,
                                  A + (size_t)mBlock * lda + k1, lda, Mtot, 128);
                if (BMODE == 0)
                    tdm_load_tile_f16(bsOff0 + (unsigned)nxt * bsStep,
                                      Bm + (size_t)nBlock * ldb + k1, ldb, Ntot, 64);
            }
        }
        // ---- wait only for step ks; step ks+1 stays in flight behind compute ----
        if (wave == 0) {
            if (ks + 1 < nk) __builtin_amdgcn_s_wait_tensorcnt(BMODE == 0 ? 2 : 1);
            else             __builtin_amdgcn_s_wait_tensorcnt(0);
        }
        __syncthreads();   // buffer `cur` ready for everyone

        // --- A fragments (ISA 16-bit A 16x32 layout) ---
        v16h a0, a1;
        {
            const int m0 = wm + (lane & 15);
            const int h8 = (lane >> 4) * 8;   // 0 or 8
#pragma unroll
            for (int e = 0; e < 8; ++e) a0[e]     = As[cur][m0 * 40 + h8 + e];
#pragma unroll
            for (int e = 0; e < 8; ++e) a0[8 + e] = As[cur][m0 * 40 + 16 + h8 + e];
            const int m1 = m0 + 16;
#pragma unroll
            for (int e = 0; e < 8; ++e) a1[e]     = As[cur][m1 * 40 + h8 + e];
#pragma unroll
            for (int e = 0; e < 8; ++e) a1[8 + e] = As[cur][m1 * 40 + 16 + h8 + e];
        }
        // --- B fragments (32x16: same K order as A: lanes 0-15 K=0-7,16-23; lanes 16-31 K=8-15,24-31) ---
        v16h b0, b1;
        {
            const int n0  = wn + (lane & 15);
            const int k16 = (lane >> 4) * 8;
#pragma unroll
            for (int e = 0; e < 16; ++e) b0[e] = Bs[cur][n0 * 40 + (e >> 3) * 16 + k16 + (e & 7)];
#pragma unroll
            for (int e = 0; e < 16; ++e) b1[e] = Bs[cur][(n0 + 16) * 40 + (e >> 3) * 16 + k16 + (e & 7)];
        }
        acc00 = __builtin_amdgcn_wmma_f32_16x16x32_f16(false, a0, false, b0,
                                                       (short)0, acc00, false, false);
        acc01 = __builtin_amdgcn_wmma_f32_16x16x32_f16(false, a0, false, b1,
                                                       (short)0, acc01, false, false);
        acc10 = __builtin_amdgcn_wmma_f32_16x16x32_f16(false, a1, false, b0,
                                                       (short)0, acc10, false, false);
        acc11 = __builtin_amdgcn_wmma_f32_16x16x32_f16(false, a1, false, b1,
                                                       (short)0, acc11, false, false);
        __syncthreads();   // done reading `cur`; it may be overwritten next iter
    }

    store_tile(acc00, mBlock + wm,      nBlock + wn,      lane, bias, alpha,
               applyMask, actGelu, accum, outF32, outF16, ldc);
    store_tile(acc01, mBlock + wm,      nBlock + wn + 16, lane, bias, alpha,
               applyMask, actGelu, accum, outF32, outF16, ldc);
    store_tile(acc10, mBlock + wm + 16, nBlock + wn,      lane, bias, alpha,
               applyMask, actGelu, accum, outF32, outF16, ldc);
    store_tile(acc11, mBlock + wm + 16, nBlock + wn + 16, lane, bias, alpha,
               applyMask, actGelu, accum, outF32, outF16, ldc);
}

// ---------------------------------------------------------------------------
// Row softmax over 4096 f16 cols, in place. One block per row.
// ---------------------------------------------------------------------------
__global__ __launch_bounds__(256) void softmax_kernel(_Float16* __restrict__ sp) {
    __shared__ float red[256];
    const int tid = threadIdx.x;
    _Float16* p = sp + (size_t)blockIdx.x * Mrows;
    float vals[16];
    float mx = -3.0e38f;
#pragma unroll
    for (int i = 0; i < 16; ++i) {
        float v = (float)p[tid + (i << 8)];
        vals[i] = v;
        mx = fmaxf(mx, v);
    }
    red[tid] = mx; __syncthreads();
    for (int s = 128; s > 0; s >>= 1) {
        if (tid < s) red[tid] = fmaxf(red[tid], red[tid + s]);
        __syncthreads();
    }
    mx = red[0]; __syncthreads();
    float sum = 0.0f;
#pragma unroll
    for (int i = 0; i < 16; ++i) { vals[i] = __expf(vals[i] - mx); sum += vals[i]; }
    red[tid] = sum; __syncthreads();
    for (int s = 128; s > 0; s >>= 1) {
        if (tid < s) red[tid] += red[tid + s];
        __syncthreads();
    }
    const float inv = 1.0f / red[0];
#pragma unroll
    for (int i = 0; i < 16; ++i) p[tid + (i << 8)] = (_Float16)(vals[i] * inv);
}

// ---------------------------------------------------------------------------
// x = LN(x + addend); writes f32 x and its f16 mirror. One block per row (D=512).
// ---------------------------------------------------------------------------
__global__ __launch_bounds__(256)
void residual_ln_kernel(float* __restrict__ x, const float* __restrict__ addend,
                        const float* __restrict__ g, const float* __restrict__ b,
                        _Float16* __restrict__ xh) {
    __shared__ float red[256];
    const int tid = threadIdx.x;
    const size_t base = (size_t)blockIdx.x * Dd;
    float v0 = x[base + tid]       + addend[base + tid];
    float v1 = x[base + tid + 256] + addend[base + tid + 256];
    red[tid] = v0 + v1; __syncthreads();
    for (int s = 128; s > 0; s >>= 1) {
        if (tid < s) red[tid] += red[tid + s];
        __syncthreads();
    }
    const float mean = red[0] * (1.0f / Dd); __syncthreads();
    const float d0 = v0 - mean, d1 = v1 - mean;
    red[tid] = d0 * d0 + d1 * d1; __syncthreads();
    for (int s = 128; s > 0; s >>= 1) {
        if (tid < s) red[tid] += red[tid + s];
        __syncthreads();
    }
    const float inv = rsqrtf(red[0] * (1.0f / Dd) + 1e-5f);
    const float o0 = d0 * inv * g[tid]       + b[tid];
    const float o1 = d1 * inv * g[tid + 256] + b[tid + 256];
    x[base + tid]        = o0;  xh[base + tid]        = (_Float16)o0;
    x[base + tid + 256]  = o1;  xh[base + tid + 256]  = (_Float16)o1;
}

// ---------------------------------------------------------------------------
// Item 0 has an empty bank: attn_out rows 0..31 come from the original input.
// ---------------------------------------------------------------------------
__global__ __launch_bounds__(256)
void select_item0_kernel(_Float16* __restrict__ xh, const _Float16* __restrict__ memh,
                         float* __restrict__ xf, const float* __restrict__ inF32) {
    int i = blockIdx.x * 256 + threadIdx.x;   // i < T*D
    xh[i] = memh[i];
    xf[i] = inF32[i];
}

// ---------------------------------------------------------------------------
// out = sigmoid(logits) * input + (1 - sigmoid) * attn_out
// ---------------------------------------------------------------------------
__global__ __launch_bounds__(256)
void gate_blend_kernel(float* __restrict__ out, const float* __restrict__ logits,
                       const float* __restrict__ inp, const float* __restrict__ attn) {
    int i = blockIdx.x * 256 + threadIdx.x;
    const float s = 1.0f / (1.0f + __expf(-logits[i]));
    out[i] = s * inp[i] + (1.0f - s) * attn[i];
}

// ---------------------------------------------------------------------------
extern "C" void kernel_launch(void* const* d_in, const int* in_sizes, int n_in,
                              void* d_out, int out_size, void* d_ws, size_t ws_size,
                              hipStream_t stream) {
    const float* inX  = (const float*)d_in[0];
    const float* Wq   = (const float*)d_in[1];  const float* bq = (const float*)d_in[2];
    const float* Wk   = (const float*)d_in[3];  const float* bk = (const float*)d_in[4];
    const float* Wv   = (const float*)d_in[5];  const float* bv = (const float*)d_in[6];
    const float* ln1g = (const float*)d_in[7];  const float* ln1b = (const float*)d_in[8];
    const float* W1   = (const float*)d_in[9];  const float* b1 = (const float*)d_in[10];
    const float* W2   = (const float*)d_in[11]; const float* b2 = (const float*)d_in[12];
    const float* ln2g = (const float*)d_in[13]; const float* ln2b = (const float*)d_in[14];
    const float* Ws   = (const float*)d_in[15]; const float* bs = (const float*)d_in[16];
    float* out = (float*)d_out;

    const int M = Mrows, D = Dd;
    const size_t MD   = (size_t)M * D;              // 2,097,152
    const size_t LDD  = (size_t)Ll * D * D;         //   524,288
    const size_t L4DD = (size_t)Ll * 4 * D * D;     // 2,097,152
    const size_t WSN  = (size_t)D * 2 * D;          //   524,288
    const size_t WH_N = 3 * LDD + 2 * L4DD + WSN;   // 6,291,456 halves

    // workspace layout (256B aligned)
    size_t off = 0;
    auto take = [&](size_t bytes) -> size_t {
        off = (off + 255) & ~(size_t)255;
        size_t o = off; off += bytes; return o;
    };
    char* w = (char*)d_ws;
    _Float16* wh   = (_Float16*)(w + take(WH_N * 2));
    _Float16* xh   = (_Float16*)(w + take(MD * 2));
    _Float16* memh = (_Float16*)(w + take(MD * 2));
    _Float16* qh   = (_Float16*)(w + take(MD * 2));
    _Float16* kh   = (_Float16*)(w + take(MD * 2));
    _Float16* vh   = (_Float16*)(w + take(MD * 2));
    _Float16* hh   = (_Float16*)(w + take((size_t)M * 4 * D * 2));
    _Float16* sp   = (_Float16*)(w + take((size_t)M * M * 2));
    float*    tf32 = (float*)   (w + take(MD * 4));
    float*    xf32 = (float*)   (w + take(MD * 4));

    _Float16* whq = wh;
    _Float16* whk = wh + LDD;
    _Float16* whv = wh + 2 * LDD;
    _Float16* wh1 = wh + 3 * LDD;
    _Float16* wh2 = wh + 3 * LDD + L4DD;
    _Float16* whs = wh + 3 * LDD + 2 * L4DD;

    // convert input + weights to f16
    cvt_f16_kernel<<<MD / 256, 256, 0, stream>>>(inX, memh);
    cvt_f16_kernel<<<LDD / 256, 256, 0, stream>>>(Wq, whq);
    cvt_f16_kernel<<<LDD / 256, 256, 0, stream>>>(Wk, whk);
    cvt_f16_kernel<<<LDD / 256, 256, 0, stream>>>(Wv, whv);
    cvt_f16_kernel<<<L4DD / 256, 256, 0, stream>>>(W1, wh1);
    cvt_f16_kernel<<<L4DD / 256, 256, 0, stream>>>(W2, wh2);
    cvt_f16_kernel<<<WSN / 256, 256, 0, stream>>>(Ws, whs);
    hipMemcpyAsync(xh, memh, MD * 2, hipMemcpyDeviceToDevice, stream);
    hipMemcpyAsync(xf32, inX, MD * 4, hipMemcpyDeviceToDevice, stream);

    auto gemmBT = [&](const _Float16* A, int lda, const _Float16* Bm, int ldb,
                      const float* bias, float alpha, int Mg, int Ng, int K,
                      int mask, int gelu, int accum,
                      float* o32, _Float16* o16, int ldc) {
        dim3 g(Ng / 64, Mg / 128);
        gemm_wmma_kernel<0><<<g, 256, 0, stream>>>(A, lda, Bm, ldb, bias, alpha, K,
                                                   mask, gelu, accum, o32, o16, ldc);
    };
    auto gemmBN = [&](const _Float16* A, int lda, const _Float16* Bm, int ldb,
                      float alpha, int Mg, int Ng, int K,
                      float* o32, _Float16* o16, int ldc) {
        dim3 g(Ng / 64, Mg / 128);
        gemm_wmma_kernel<1><<<g, 256, 0, stream>>>(A, lda, Bm, ldb, nullptr, alpha, K,
                                                   0, 0, 0, o32, o16, ldc);
    };

    const float scale = 0.044194173824159216f;   // 1/sqrt(512)

    for (int l = 0; l < Ll; ++l) {
        const size_t wo = (size_t)l * D * D, w1o = (size_t)l * 4 * D * D;
        // q = x @ Wq^T + bq ; k,v from memory bank (original input)
        gemmBT(xh,   D, whq + wo, D, bq + l * D, 1.0f, M, D, D, 0, 0, 0, nullptr, qh, D);
        gemmBT(memh, D, whk + wo, D, bk + l * D, 1.0f, M, D, D, 0, 0, 0, nullptr, kh, D);
        gemmBT(memh, D, whv + wo, D, bv + l * D, 1.0f, M, D, D, 0, 0, 0, nullptr, vh, D);
        // s = scale * q @ k^T  with item-visibility mask
        gemmBT(qh, D, kh, D, nullptr, scale, M, M, D, 1, 0, 0, nullptr, sp, M);
        softmax_kernel<<<M, 256, 0, stream>>>(sp);
        // attn = p @ v
        gemmBN(sp, M, vh, D, 1.0f, M, D, M, tf32, nullptr, D);
        residual_ln_kernel<<<M, 256, 0, stream>>>(xf32, tf32, ln1g + l * D, ln1b + l * D, xh);
        // MLP
        gemmBT(xh, D, wh1 + w1o, D, b1 + l * 4 * D, 1.0f, M, 4 * D, D, 0, 1, 0,
               nullptr, hh, 4 * D);
        gemmBT(hh, 4 * D, wh2 + w1o, 4 * D, b2 + l * D, 1.0f, M, D, 4 * D, 0, 0, 0,
               tf32, nullptr, D);
        residual_ln_kernel<<<M, 256, 0, stream>>>(xf32, tf32, ln2g + l * D, ln2b + l * D, xh);
    }

    // attn_out rows of item 0 = original input
    select_item0_kernel<<<(Tt * Dd) / 256, 256, 0, stream>>>(xh, memh, xf32, inX);

    // gate: logits = [input, attn_out] @ Ws^T + bs   (Ws stored [D, 2D])
    gemmBT(memh, D, whs,     2 * D, bs, 1.0f, M, D, D, 0, 0, 0, tf32, nullptr, D);
    gemmBT(xh,   D, whs + D, 2 * D, nullptr, 1.0f, M, D, D, 0, 0, 1, tf32, nullptr, D);

    gate_blend_kernel<<<MD / 256, 256, 0, stream>>>(out, tf32, inX, xf32);
}
